// DwiConv3d_52063593562462
// MI455X (gfx1250) — matrix operand order repaired and checked by reading, not compiled, not run on hardware
//
#include <hip/hip_runtime.h>

typedef __attribute__((ext_vector_type(16))) __bf16 v16bf;
typedef __attribute__((ext_vector_type(8)))  __bf16 v8bf;
typedef __attribute__((ext_vector_type(8)))  float  v8f;

#define NB    4      // batch
#define CIN   36     // input channels (M*N)
#define CPAD  40     // padded c so (d,c) unit = 80B (16B aligned)
#define DIM   64     // D=H=W
#define OCH   8      // output channels
#define ODIM  62     // output spatial dim
#define TW    16     // ow tile (WMMA M)
#define XW    18     // input tile width (16 + 2 halo)
// sX: [h:4][w:18][d:4][c:40] bf16 (+64 elem tail guard for k-padding reads)
#define SX_ELEMS (4 * XW * 4 * CPAD + 64)     // 11584 elems = 23168 B (16B mult)
// sB: [tap:9][step:4][chunk:2][lane:32][8] bf16
#define SB_ELEMS (9 * 4 * 512)                // 18432 elems = 36864 B
#define NWVAL (OCH * 972)                     // 7776 nonzero weight entries

__global__ __launch_bounds__(128)
void dwiconv3d_wmma_bf16(const float* __restrict__ x,
                         const float* __restrict__ w,
                         float* __restrict__ out)
{
    __shared__ __align__(16) __bf16 sX[SX_ELEMS];
    __shared__ __align__(16) __bf16 sB[SB_ELEMS];

    const int tid  = threadIdx.x;
    const int lane = tid & 31;
    const int wave = tid >> 5;

    const int ow0 = blockIdx.x * TW;          // 0,16,32,48
    const int odt = blockIdx.y / 31;
    const int oht = blockIdx.y % 31;
    const int od0 = odt * 2;
    const int oh0 = oht * 2;
    const int b   = blockIdx.z;

    // ---- Phase P: prefetch all 576 input-tile rows (global_prefetch_b8),
    //      overlapping HBM latency with the zero + B-expansion ALU work. ----
    for (int idx = tid; idx < CIN * 4 * 4; idx += 128) {
        int h = idx % 4;
        int d = (idx / 4) % 4;
        int c = idx / 16;
        const float* p =
            &x[(((size_t)(b * CIN + c) * DIM + (od0 + d)) * DIM + (oh0 + h)) * DIM + ow0];
        __builtin_prefetch(p, 0, 0);
    }

    // ---- Phase 0: zero both LDS tiles with int4 stores ----
    {
        int4 zz = {0, 0, 0, 0};
        int4* zx = (int4*)sX;                 // 1448 int4
        for (int i = tid; i < (int)(SX_ELEMS * 2 / 16); i += 128) zx[i] = zz;
        int4* zb = (int4*)sB;                 // 2304 int4
        for (int i = tid; i < (int)(SB_ELEMS * 2 / 16); i += 128) zb[i] = zz;
    }
    __syncthreads();

    // ---- Phase 0b: scatter the 7776 real weights into B fragments.
    //      Coalesced reads (consecutive tid -> consecutive kw).
    //      B lane map: lane L holds N = L&15; elem e -> K = step*32 + 16*(L>=16) + e.
    //      K packs (kd,c) as kd*40 + c. Chunk-major: [tap*4+step][chunk][lane][8]. ----
    for (int idx = tid; idx < NWVAL; idx += 128) {
        int o  = idx / 972;
        int r  = idx % 972;
        int c  = r / 27;
        int rr = r % 27;
        int kd = rr / 9;
        int kh = (rr % 9) / 3;
        int kw = rr % 3;
        int tap = kh * 3 + kw;
        int kk  = kd * CPAD + c;
        int step = kk >> 5;
        int el   = kk & 31;
        int l  = (el < 16) ? o : (o + 16);
        int e  = (el < 16) ? el : (el - 16);
        int c2 = e >> 3;
        int j  = e & 7;
        sB[(tap * 4 + step) * 512 + c2 * 256 + l * 8 + j] = (__bf16)w[idx];
    }

    // ---- Phase 1: input halo tile, w-innermost global reads (coalesced) ----
    for (int idx = tid; idx < CIN * 4 * 4 * XW; idx += 128) {
        int ww = idx % XW;
        int h  = (idx / XW) % 4;
        int d  = (idx / (XW * 4)) % 4;
        int c  = idx / (XW * 4 * 4);
        int wg = ow0 + ww;
        float v = 0.f;
        if (wg < DIM)
            v = x[(((size_t)(b * CIN + c) * DIM + (od0 + d)) * DIM + (oh0 + h)) * DIM + wg];
        sX[((h * XW + ww) * 4 + d) * CPAD + c] = (__bf16)v;
    }
    __syncthreads();

    // ---- Phase 2: 36 WMMAs, each fed by 4x ds_load_b128 ----
    const int dd = wave >> 1;                 // wave -> (od, oh) within 2x2
    const int hh = wave & 1;
    const int od = od0 + dd;
    const int oh = oh0 + hh;
    const int m  = lane & 15;                 // A row = ow0 + m
    const int hi16 = (lane >= 16) ? 32 : 0;   // A hi-half K offset (bytes)

    const char* sXb = (const char*)sX;
    const char* sBb = (const char*)sB;

    v8f acc = {};
    #pragma unroll
    for (int kh = 0; kh < 3; ++kh) {
        #pragma unroll
        for (int kw = 0; kw < 3; ++kw) {
            const int tap   = kh * 3 + kw;
            // 16B-aligned: (d,c) unit stride is 80B
            const int abase = (((hh + kh) * XW + (m + kw)) * 4 + dd) * (CPAD * 2) + hi16;
            #pragma unroll
            for (int step = 0; step < 4; ++step) {
                v8bf a0 = *(const v8bf*)(sXb + abase + step * 64);
                v8bf a1 = *(const v8bf*)(sXb + abase + step * 64 + 16);
                v8bf b0 = *(const v8bf*)(sBb + (tap * 4 + step) * 1024 + lane * 16);
                v8bf b1 = *(const v8bf*)(sBb + (tap * 4 + step) * 1024 + 512 + lane * 16);
                v16bf a  = __builtin_shufflevector(a0, a1,
                            0,1,2,3,4,5,6,7,8,9,10,11,12,13,14,15);
                v16bf bf = __builtin_shufflevector(b0, b1,
                            0,1,2,3,4,5,6,7,8,9,10,11,12,13,14,15);
                acc = __builtin_amdgcn_wmma_f32_16x16x32_bf16(
                          false, a, false, bf, (short)0, acc, false, false);
            }
        }
    }

    // ---- Phase 3: C/D layout: lane -> N = lane&15; VGPR r -> M = r + 8*(lane>=16)
    const int n  = lane & 15;
    const int mh = (lane >= 16) ? 8 : 0;
    if (n < OCH) {
        #pragma unroll
        for (int r = 0; r < 8; ++r) {
            int ow = ow0 + r + mh;
            if (ow < ODIM)
                out[((((size_t)b * OCH + n) * ODIM + od) * ODIM + oh) * ODIM + ow] = acc[r];
        }
    }
}

extern "C" void kernel_launch(void* const* d_in, const int* in_sizes, int n_in,
                              void* d_out, int out_size, void* d_ws, size_t ws_size,
                              hipStream_t stream)
{
    const float* x = (const float*)d_in[0];   // [4, 36, 64,64,64] f32
    const float* w = (const float*)d_in[1];   // [8, 36, 3,3,3]    f32
    float* out = (float*)d_out;               // [4, 8, 62,62,62]  f32

    dim3 grid(4, 31 * 31, NB);  // ow tiles x (od,oh)/2x2 tiles x batch
    dim3 block(128);            // 4 wave32s
    dwiconv3d_wmma_bf16<<<grid, block, 0, stream>>>(x, w, out);
}
